// net_86225763434796
// MI455X (gfx1250) — matrix operand order repaired and checked by reading, not compiled, not run on hardware
//
#include <hip/hip_runtime.h>

typedef __attribute__((ext_vector_type(16))) _Float16 v16h;
typedef __attribute__((ext_vector_type(8)))  _Float16 v8h;
typedef __attribute__((ext_vector_type(8)))  float    v8f;

#define N_NODES 100000
#define DSIG    16
#define EDIM    3
#define KNB     32
#define HID     64
#define NOUT    3
#define FEAT    208
#define FPAD    224      // 7 * 32 (WMMA K-steps)
#define KT      7
#define NT      4        // 64 / 16

// workspace layout (bytes); all offsets 32B-aligned
#define WS_XD   0u
#define WS_G1   (N_NODES * DSIG * 4u)                      //  6,400,000
#define WS_F    (WS_G1 + N_NODES * 48u * 4u)               // 25,600,000
#define WS_W1B  (WS_F + N_NODES * FPAD * 2u)               // 70,400,000
// total = 70,428,672 bytes

// ---------------------------------------------------------------------------
// K0: swizzle W1 [208,64] f32 -> B-operand f16 tiles, K zero-padded to 224.
// B layout (16x16x32 f16): lane<16: col N=lane, 16 halves = K {0..7,16..23} of k-tile;
//                          lane>=16: col N=lane-16, halves = K {8..15,24..31}.
// Tile (kt,nt) stored at W1B + (kt*4+nt)*512 halves, lane-major, 16 halves/lane.
// ---------------------------------------------------------------------------
__global__ __launch_bounds__(256) void k_prep_w1(const float* __restrict__ W1,
                                                 _Float16* __restrict__ W1B) {
    int t = blockIdx.x * blockDim.x + threadIdx.x;
    if (t >= KT * NT * 32) return;
    int lane = t & 31;
    int tile = t >> 5;
    int nt = tile & 3, kt = tile >> 2;
    int n = nt * 16 + (lane & 15);
    int kbase = kt * 32 + ((lane < 16) ? 0 : 8);
    _Float16* dst = W1B + tile * 512 + lane * 16;
#pragma unroll
    for (int h = 0; h < 16; ++h) {
        int k = kbase + (h & 7) + ((h >> 3) << 4);
        float v = (k < FEAT) ? W1[k * HID + n] : 0.0f;
        dst[h] = (_Float16)v;
    }
}

// ---------------------------------------------------------------------------
// K1: diffusion  xd = 0.5*x + 0.5*mean_K x[nb];  also F[:,0:16] and pad cols.
// ---------------------------------------------------------------------------
__global__ __launch_bounds__(256) void k_diffuse(const float* __restrict__ x,
                                                 const int* __restrict__ nb,
                                                 float* __restrict__ xd,
                                                 _Float16* __restrict__ F) {
    int n = blockIdx.x * blockDim.x + threadIdx.x;
    if (n >= N_NODES) return;
    float acc[DSIG];
#pragma unroll
    for (int c = 0; c < DSIG; ++c) acc[c] = 0.0f;
    const int* nbr = nb + n * KNB;
    for (int k = 0; k < KNB; ++k) {
        int idx = nbr[k];
        const float4* xr = (const float4*)(x + (long)idx * DSIG);
#pragma unroll
        for (int j = 0; j < 4; ++j) {
            float4 q = xr[j];
            acc[4 * j + 0] += q.x; acc[4 * j + 1] += q.y;
            acc[4 * j + 2] += q.z; acc[4 * j + 3] += q.w;
        }
    }
    const float* xs = x + (long)n * DSIG;
    float* xo = xd + (long)n * DSIG;
    _Float16* Fr = F + (long)n * FPAD;
#pragma unroll
    for (int c = 0; c < DSIG; ++c) {
        float v = 0.5f * xs[c] + acc[c] * (0.5f / (float)KNB);
        xo[c] = v;
        Fr[c] = (_Float16)v;
    }
    // zero the K padding cols 208..223
    v8h z = {};
    *(v8h*)(Fr + 208) = z;
    *(v8h*)(Fr + 216) = z;
}

// ---------------------------------------------------------------------------
// K2: first directional gradient  g1[n, e*16+c] = sum_k kern[e,n,k]*xd[nb,c]
// ---------------------------------------------------------------------------
__global__ __launch_bounds__(256) void k_grad1(const float* __restrict__ xd,
                                               const int* __restrict__ nb,
                                               const float* __restrict__ kern,
                                               float* __restrict__ g1,
                                               _Float16* __restrict__ F) {
    int n = blockIdx.x * blockDim.x + threadIdx.x;
    if (n >= N_NODES) return;
    float a[EDIM][DSIG];
#pragma unroll
    for (int e = 0; e < EDIM; ++e)
#pragma unroll
        for (int c = 0; c < DSIG; ++c) a[e][c] = 0.0f;
    const int* nbr = nb + n * KNB;
    const long nk = (long)N_NODES * KNB;
    for (int k = 0; k < KNB; ++k) {
        int idx = nbr[k];
        float w[EDIM];
#pragma unroll
        for (int e = 0; e < EDIM; ++e) w[e] = kern[(long)e * nk + (long)n * KNB + k];
        const float4* gr = (const float4*)(xd + (long)idx * DSIG);
        float r[DSIG];
#pragma unroll
        for (int j = 0; j < 4; ++j) {
            float4 q = gr[j];
            r[4 * j + 0] = q.x; r[4 * j + 1] = q.y;
            r[4 * j + 2] = q.z; r[4 * j + 3] = q.w;
        }
#pragma unroll
        for (int e = 0; e < EDIM; ++e)
#pragma unroll
            for (int c = 0; c < DSIG; ++c) a[e][c] += w[e] * r[c];
    }
    float* go = g1 + (long)n * 48;
    _Float16* Fr = F + (long)n * FPAD;
#pragma unroll
    for (int e = 0; e < EDIM; ++e)
#pragma unroll
        for (int c = 0; c < DSIG; ++c) {
            float v = a[e][c];
            go[e * DSIG + c] = v;
            Fr[DSIG + e * DSIG + c] = (_Float16)v;
        }
}

// ---------------------------------------------------------------------------
// K3: second gradient  F[n, 64 + e*48 + j] = sum_k kern[e,n,k]*g1[nb, j]
// one thread per (node, e)
// ---------------------------------------------------------------------------
__global__ __launch_bounds__(256) void k_grad2(const float* __restrict__ g1,
                                               const int* __restrict__ nb,
                                               const float* __restrict__ kern,
                                               _Float16* __restrict__ F) {
    int t = blockIdx.x * blockDim.x + threadIdx.x;
    if (t >= N_NODES * EDIM) return;
    int n = t / EDIM;
    int e = t - n * EDIM;
    float acc[48];
#pragma unroll
    for (int j = 0; j < 48; ++j) acc[j] = 0.0f;
    const int* nbr = nb + n * KNB;
    const float* kw = kern + (long)e * N_NODES * KNB + (long)n * KNB;
    for (int k = 0; k < KNB; ++k) {
        int idx = nbr[k];
        float w = kw[k];
        const float4* gr = (const float4*)(g1 + (long)idx * 48);
#pragma unroll
        for (int j = 0; j < 12; ++j) {
            float4 q = gr[j];
            acc[4 * j + 0] += w * q.x; acc[4 * j + 1] += w * q.y;
            acc[4 * j + 2] += w * q.z; acc[4 * j + 3] += w * q.w;
        }
    }
    _Float16* Fr = F + (long)n * FPAD + 64 + e * 48;
#pragma unroll
    for (int j = 0; j < 48; ++j) Fr[j] = (_Float16)acc[j];
}

// ---------------------------------------------------------------------------
// K4: WMMA MLP head.  Each wave owns a 16-node tile:
//   h = relu(F[16x224] @ W1[224x64] + b1)  via 7x4 v_wmma_f32_16x16x32_f16
//   out = h @ W2[64x3] + b2                via LDS + VALU dot (tiny)
// ---------------------------------------------------------------------------
__global__ __launch_bounds__(256) void k_mlp(const _Float16* __restrict__ F,
                                             const _Float16* __restrict__ W1B,
                                             const float* __restrict__ b1,
                                             const float* __restrict__ W2,
                                             const float* __restrict__ b2,
                                             float* __restrict__ out) {
    __shared__ float hshm[8][16][HID];
    const int lane = threadIdx.x & 31;
    const int wv = threadIdx.x >> 5;
    const int tile = blockIdx.x * 8 + wv;
    const int m0 = tile * 16;
    const int ncol = lane & 15;

    if (m0 < N_NODES) {
        // init accumulators with broadcast bias (C/D layout: col = 16*nt + ncol)
        float bv0 = b1[0 * 16 + ncol], bv1 = b1[1 * 16 + ncol];
        float bv2 = b1[2 * 16 + ncol], bv3 = b1[3 * 16 + ncol];
        v8f c0, c1, c2, c3;
#pragma unroll
        for (int i = 0; i < 8; ++i) { c0[i] = bv0; c1[i] = bv1; c2[i] = bv2; c3[i] = bv3; }

        // A-operand: lane<16 holds row m0+lane, K {0..7,16..23} of the k-tile;
        //            lane>=16 holds row m0+lane-16, K {8..15,24..31}
        const _Float16* arow = F + (long)(m0 + ncol) * FPAD;
        const int aoff = (lane < 16) ? 0 : 8;
        const v16h* bbase = (const v16h*)W1B;

#pragma unroll
        for (int kt = 0; kt < KT; ++kt) {
            const v8h* ap = (const v8h*)(arow + kt * 32 + aoff);
            v8h lo = ap[0];   // halves base..base+7
            v8h hi = ap[2];   // halves base+16..base+23
            v16h a = __builtin_shufflevector(lo, hi, 0, 1, 2, 3, 4, 5, 6, 7,
                                                     8, 9, 10, 11, 12, 13, 14, 15);
            v16h bt0 = bbase[(kt * 4 + 0) * 32 + lane];
            v16h bt1 = bbase[(kt * 4 + 1) * 32 + lane];
            v16h bt2 = bbase[(kt * 4 + 2) * 32 + lane];
            v16h bt3 = bbase[(kt * 4 + 3) * 32 + lane];
            c0 = __builtin_amdgcn_wmma_f32_16x16x32_f16(false, a, false, bt0, (short)0, c0, false, false);
            c1 = __builtin_amdgcn_wmma_f32_16x16x32_f16(false, a, false, bt1, (short)0, c1, false, false);
            c2 = __builtin_amdgcn_wmma_f32_16x16x32_f16(false, a, false, bt2, (short)0, c2, false, false);
            c3 = __builtin_amdgcn_wmma_f32_16x16x32_f16(false, a, false, bt3, (short)0, c3, false, false);
        }

        // relu + spill to LDS (C/D layout: lane<16 -> rows r, lane>=16 -> rows r+8)
        const int mbase = (lane < 16) ? 0 : 8;
#pragma unroll
        for (int r = 0; r < 8; ++r) {
            hshm[wv][mbase + r][0 * 16 + ncol] = fmaxf(c0[r], 0.0f);
            hshm[wv][mbase + r][1 * 16 + ncol] = fmaxf(c1[r], 0.0f);
            hshm[wv][mbase + r][2 * 16 + ncol] = fmaxf(c2[r], 0.0f);
            hshm[wv][mbase + r][3 * 16 + ncol] = fmaxf(c3[r], 0.0f);
        }
    }
    __syncthreads();

    if (m0 < N_NODES) {
        // second layer: 16 rows x 3 outputs = 48 dot products of length 64
        for (int p = lane; p < 16 * NOUT; p += 32) {
            int o = p / 16;
            int r = p - o * 16;
            float acc = b2[o];
            const float* hr = &hshm[wv][r][0];
#pragma unroll 8
            for (int h = 0; h < HID; ++h) acc += hr[h] * W2[h * NOUT + o];
            out[(long)(m0 + r) * NOUT + o] = acc;
        }
    }
}

// ---------------------------------------------------------------------------
extern "C" void kernel_launch(void* const* d_in, const int* in_sizes, int n_in,
                              void* d_out, int out_size, void* d_ws, size_t ws_size,
                              hipStream_t stream) {
    const float* x    = (const float*)d_in[0];
    const int*   nb   = (const int*)d_in[1];
    const float* kern = (const float*)d_in[2];
    const float* W1   = (const float*)d_in[3];
    const float* b1   = (const float*)d_in[4];
    const float* W2   = (const float*)d_in[5];
    const float* b2   = (const float*)d_in[6];
    float* out = (float*)d_out;

    char* ws = (char*)d_ws;
    float*    xd  = (float*)(ws + WS_XD);
    float*    g1  = (float*)(ws + WS_G1);
    _Float16* F   = (_Float16*)(ws + WS_F);
    _Float16* W1B = (_Float16*)(ws + WS_W1B);

    k_prep_w1<<<4, 256, 0, stream>>>(W1, W1B);
    k_diffuse<<<(N_NODES + 255) / 256, 256, 0, stream>>>(x, nb, xd, F);
    k_grad1<<<(N_NODES + 255) / 256, 256, 0, stream>>>(xd, nb, kern, g1, F);
    k_grad2<<<(N_NODES * EDIM + 255) / 256, 256, 0, stream>>>(g1, nb, kern, F);

    int tiles = (N_NODES + 15) / 16;          // 6250
    int blocks = (tiles + 7) / 8;             // 782
    k_mlp<<<blocks, 256, 0, stream>>>(F, W1B, b1, W2, b2, out);
}
